// Lion_Attention_37125697307265
// MI455X (gfx1250) — matrix operand order repaired and checked by reading, not compiled, not run on hardware
//
#include <hip/hip_runtime.h>
#include <hip/hip_bf16.h>

// ---------------- types ----------------
typedef __attribute__((ext_vector_type(16))) __bf16        v16bf;
typedef __attribute__((ext_vector_type(8)))  float          v8f;
typedef __attribute__((ext_vector_type(8)))  unsigned short ushort8;
typedef __attribute__((ext_vector_type(4)))  float          float4v;
typedef __attribute__((ext_vector_type(4)))  unsigned short ushort4v;

union FragU { ushort8 u[2]; v16bf bf; };

__device__ __forceinline__ unsigned short f32_to_bf16(float f) {
    union { float f; unsigned u; } v; v.f = f;
    unsigned r = v.u + 0x7FFFu + ((v.u >> 16) & 1u);   // round to nearest even
    return (unsigned short)(r >> 16);
}
__device__ __forceinline__ float bf16_to_f32(unsigned short s) {
    union { unsigned u; float f; } v; v.u = ((unsigned)s) << 16;
    return v.f;
}
__device__ __forceinline__ v8f zero8() { v8f z = {0.f,0.f,0.f,0.f,0.f,0.f,0.f,0.f}; return z; }

// LDS byte offset of a __shared__ pointer (aperture rule: offset = addr[31:0]).
__device__ __forceinline__ unsigned lds_off(const void* p) {
    return (unsigned)(unsigned long long)p;
}
// CDNA5 async global->LDS copy, 16 bytes per lane (GLOBAL_LOAD_ASYNC_TO_LDS_B128, GV mode).
__device__ __forceinline__ void async_copy_b128(unsigned lds, const void* gptr) {
    asm volatile("global_load_async_to_lds_b128 %0, %1, off"
                 :: "v"(lds), "v"((unsigned long long)gptr)
                 : "memory");
}
__device__ __forceinline__ void wait_async0() {
    asm volatile("s_wait_asynccnt 0x0" ::: "memory");
}

// A fragment: 16x32 bf16 tile stored [row][k] (row stride = ldk ushorts).
__device__ __forceinline__ v16bf load_a_frag(const unsigned short* base, int ldk) {
    const int l   = threadIdx.x & 31;
    const int row = l & 15;
    const int kb  = (l >> 4) * 8;
    FragU f;
    f.u[0] = *(const ushort8*)(base + row * ldk + kb);
    f.u[1] = *(const ushort8*)(base + row * ldk + 16 + kb);
    return f.bf;
}
// B fragment: 32x16 bf16 tile (KxN) stored as [n][k] (n stride = ldk ushorts).
__device__ __forceinline__ v16bf load_b_frag(const unsigned short* base, int ldk) {
    const int l  = threadIdx.x & 31;
    const int n  = l & 15;
    const int ks = (l >> 4) * 8;
    FragU f;
    f.u[0] = *(const ushort8*)(base + n * ldk + ks);
    f.u[1] = *(const ushort8*)(base + n * ldk + 16 + ks);
    return f.bf;
}
__device__ __forceinline__ v8f wmma_bf16(v16bf a, v16bf b, v8f c) {
    return __builtin_amdgcn_wmma_f32_16x16x32_bf16(false, a, false, b, (short)0, c, false, false);
}

// ---------------- constants ----------------
#define Bb   2
#define Nn   1024
#define Cc   768
#define Hh   12
#define Dd   64
#define BH   (Bb*Hh)          // 24
#define KTILES (Cc / 32)      // 24
#define SCALE 0.125f          // 64^-0.5

// ================= kernel 0: f32 -> bf16 bulk convert =================
__global__ __launch_bounds__(256) void cvt_bf16_kernel(const float* __restrict__ in,
                                                       unsigned short* __restrict__ out,
                                                       int n4) {
    const int i = blockIdx.x * 256 + threadIdx.x;
    if (i < n4) {
        float4v f = ((const float4v*)in)[i];
        ushort4v o;
        o[0] = f32_to_bf16(f[0]); o[1] = f32_to_bf16(f[1]);
        o[2] = f32_to_bf16(f[2]); o[3] = f32_to_bf16(f[3]);
        ((ushort4v*)out)[i] = o;
    }
}

// ================= kernel 1: gates + cumsum(log a) =================
__global__ __launch_bounds__(256) void gates_kernel(const float* __restrict__ X,
                                                    const float* __restrict__ Wa,
                                                    const float* __restrict__ ba,
                                                    float* __restrict__ cws) {
    __shared__ float av[Nn];
    __shared__ float cs[Nn + 1];
    const int bh = blockIdx.x;
    const int b = bh / Hh, h = bh % Hh;
    const float* xb = X + (size_t)b * Nn * Cc;
    const float* w  = Wa + h * Cc;
    for (int n = threadIdx.x; n < Nn; n += 256) {
        const float* xr = xb + (size_t)n * Cc;
        float z = ba[h];
        for (int k = 0; k < Cc; ++k) z += xr[k] * w[k];
        av[n] = 1.f - 1.f / (1.f + __expf(-z));
    }
    __syncthreads();
    if (threadIdx.x == 0) {
        float run = 0.f;
        cs[0] = 0.f;
        for (int n = 0; n < Nn; ++n) {
            float g = (n == 0) ? 1.f : av[n];
            run += __logf(g);
            cs[n + 1] = run;
        }
    }
    __syncthreads();
    for (int n = threadIdx.x; n < Nn + 1; n += 256) cws[bh * (Nn + 1) + n] = cs[n];
}

// ================= GEMM core: double-buffered async staging =================
// block: 128 thr (4 waves), tile 64x64, wave tile 32x32, K step 32.
// Pipeline per k-step: wait(asynccnt 0) -> barrier -> issue k+1 (other buffer) -> wmma(k).
// MODE 0: qkv epilogue (scatter bf16 [3,B,H,N,D]); MODE 1: proj epilogue (+bias, f32 out)
template <int MODE>
__global__ __launch_bounds__(128) void gemm_kernel(const unsigned short* __restrict__ A,
                                                   const unsigned short* __restrict__ Wb,
                                                   const float* __restrict__ bias,
                                                   unsigned short* __restrict__ outb,
                                                   float* __restrict__ outf) {
    __shared__ __align__(16) unsigned short As[2][64 * 32];   // [m][k]
    __shared__ __align__(16) unsigned short Bs[2][64 * 32];   // [n][k]
    const int t = threadIdx.x, wave = t >> 5, lane = t & 31;
    const int m0 = blockIdx.y * 64, n0 = blockIdx.x * 64;
    const int wm = (wave & 1) * 32, wn = (wave >> 1) * 32;
    const unsigned As_o = lds_off(As), Bs_o = lds_off(Bs);
    v8f acc[2][2] = { {zero8(), zero8()}, {zero8(), zero8()} };

    // chunk geometry: 64 rows x 32 cols = 256 16B chunks, 2 per thread
    const int r0 = t >> 2, cc0 = (t & 3) * 8;           // chunk t
    const int r1 = (t + 128) >> 2, cc1 = cc0;           // chunk t+128
    const unsigned l0 = (unsigned)(r0 * 32 + cc0) * 2u;
    const unsigned l1 = (unsigned)(r1 * 32 + cc1) * 2u;

    // prologue: stage k-tile 0 into buffer 0
    {
        async_copy_b128(As_o + l0, A  + (size_t)(m0 + r0) * Cc + cc0);
        async_copy_b128(Bs_o + l0, Wb + (size_t)(n0 + r0) * Cc + cc0);
        async_copy_b128(As_o + l1, A  + (size_t)(m0 + r1) * Cc + cc1);
        async_copy_b128(Bs_o + l1, Wb + (size_t)(n0 + r1) * Cc + cc1);
    }
    for (int kt = 0; kt < KTILES; ++kt) {
        wait_async0();
        __syncthreads();
        if (kt + 1 < KTILES) {   // fill the other buffer while this one is consumed
            const unsigned bo = (unsigned)(((kt + 1) & 1) * 64 * 32 * 2);
            const int kk = (kt + 1) * 32;
            async_copy_b128(As_o + bo + l0, A  + (size_t)(m0 + r0) * Cc + kk + cc0);
            async_copy_b128(Bs_o + bo + l0, Wb + (size_t)(n0 + r0) * Cc + kk + cc0);
            async_copy_b128(As_o + bo + l1, A  + (size_t)(m0 + r1) * Cc + kk + cc1);
            async_copy_b128(Bs_o + bo + l1, Wb + (size_t)(n0 + r1) * Cc + kk + cc1);
        }
        const unsigned short* as = As[kt & 1];
        const unsigned short* bs = Bs[kt & 1];
        v16bf a0 = load_a_frag(as + (wm + 0) * 32, 32);
        v16bf a1 = load_a_frag(as + (wm + 16) * 32, 32);
        v16bf b0 = load_b_frag(bs + (wn + 0) * 32, 32);
        v16bf b1 = load_b_frag(bs + (wn + 16) * 32, 32);
        acc[0][0] = wmma_bf16(a0, b0, acc[0][0]);
        acc[0][1] = wmma_bf16(a0, b1, acc[0][1]);
        acc[1][0] = wmma_bf16(a1, b0, acc[1][0]);
        acc[1][1] = wmma_bf16(a1, b1, acc[1][1]);
    }
    const int hi = lane >> 4, col = lane & 15;
    for (int mi = 0; mi < 2; ++mi)
        for (int ni = 0; ni < 2; ++ni) {
            const int ng = n0 + wn + ni * 16 + col;
            if (MODE == 0) {
                const int sec = ng / Cc, rem = ng % Cc;
                const int h = rem >> 6, d = rem & 63;
                for (int r = 0; r < 8; ++r) {
                    const int mg = m0 + wm + mi * 16 + r + 8 * hi;
                    const int b = mg >> 10, tok = mg & 1023;
                    outb[((size_t)((sec * Bb + b) * Hh + h) * Nn + tok) * Dd + d] =
                        f32_to_bf16(acc[mi][ni][r]);
                }
            } else {
                const float bv = bias[ng];
                for (int r = 0; r < 8; ++r) {
                    const int mg = m0 + wm + mi * 16 + r + 8 * hi;
                    outf[(size_t)mg * Cc + ng] = acc[mi][ni][r] + bv;
                }
            }
        }
}

// ================= kernel 3: silu(z)+0.5, L2-normalize over D (q,k in place) =========
__global__ __launch_bounds__(256) void qknorm_kernel(unsigned short* __restrict__ qkv) {
    const int row  = blockIdx.x * 8 + (threadIdx.x >> 5);
    const int lane = threadIdx.x & 31;
    unsigned short* p = qkv + (size_t)row * Dd;
    float x0 = bf16_to_f32(p[lane]);
    float x1 = bf16_to_f32(p[lane + 32]);
    float r0 = x0 / (1.f + __expf(-x0)) + 0.5f;
    float r1 = x1 / (1.f + __expf(-x1)) + 0.5f;
    float ss = r0 * r0 + r1 * r1;
    for (int m = 16; m > 0; m >>= 1) ss += __shfl_xor(ss, m, 32);
    const float inv = rsqrtf(ss);
    p[lane]      = f32_to_bf16(r0 * inv);
    p[lane + 32] = f32_to_bf16(r1 * inv);
}

// ================= kernel 4: fused attention (bf16 out) =================
__global__ __launch_bounds__(256) void attn_kernel(const unsigned short* __restrict__ qkv,
                                                   const float* __restrict__ cws,
                                                   unsigned short* __restrict__ obf) {
    __shared__ __align__(16) unsigned short Qs[64 * 64];  // [i][d]
    __shared__ __align__(16) unsigned short Ks[32 * 64];  // [j][d]
    __shared__ __align__(16) unsigned short Vs[64 * 32];  // [d][j] (transposed)
    __shared__ __align__(16) unsigned short Ps[64 * 32];  // [i][j]
    __shared__ float rowsum[64];

    const int t = threadIdx.x, wave = t >> 5, lane = t & 31, hi = lane >> 4;
    const int bh = blockIdx.x;
    const int i0 = blockIdx.y * 64;
    const unsigned short* qb = qkv + ((size_t)(0 * BH + bh)) * Nn * Dd;
    const unsigned short* kb = qkv + ((size_t)(1 * BH + bh)) * Nn * Dd;
    const unsigned short* vb = qkv + ((size_t)(2 * BH + bh)) * Nn * Dd;
    const float* cg = cws + bh * (Nn + 1);
    const unsigned Qs_o = lds_off(Qs), Ks_o = lds_off(Ks);

    if (t < 64) rowsum[t] = 0.f;
    // async-stage Q tile (8 KB = 512 chunks, 2 per thread)
    async_copy_b128(Qs_o + t * 16u,         qb + (size_t)i0 * Dd + t * 8);
    async_copy_b128(Qs_o + (t + 256) * 16u, qb + (size_t)i0 * Dd + (t + 256) * 8);

    const int rg  = (wave & 3) * 16;
    const int sc  = (wave >> 2) * 16;
    const int ocg = (wave >> 2) * 32;
    float ci[8], ci1[8];
    for (int r = 0; r < 8; ++r) {
        const int ig = i0 + rg + r + 8 * hi;
        ci[r]  = cg[ig];
        ci1[r] = cg[ig + 1];
    }
    v8f o0 = zero8(), o1 = zero8();

    for (int j0 = 0; j0 < Nn; j0 += 32) {
        // async K tile (4 KB = 256 chunks, 1 per thread)
        async_copy_b128(Ks_o + t * 16u, kb + (size_t)j0 * Dd + t * 8);
        if (j0 + 32 < Nn) __builtin_prefetch(kb + (size_t)(j0 + 32) * Dd + t * 8, 0, 3);
        // V transpose (overlaps async K): b64 global load + 4 b16 LDS stores per chunk
        #pragma unroll
        for (int c = 0; c < 2; ++c) {
            const int ch = t + c * 256;            // 512 chunks of 4 elements
            const int jj = ch & 31;
            const int d4 = (ch >> 5) * 4;
            ushort4v vv = *(const ushort4v*)(vb + (size_t)(j0 + jj) * Dd + d4);
            Vs[(d4 + 0) * 32 + jj] = vv[0];
            Vs[(d4 + 1) * 32 + jj] = vv[1];
            Vs[(d4 + 2) * 32 + jj] = vv[2];
            Vs[(d4 + 3) * 32 + jj] = vv[3];
        }
        wait_async0();
        __syncthreads();

        // S = Q K^T  (16x16 per wave, K = D = 64 -> two wmma)
        v8f s = zero8();
        for (int dk = 0; dk < Dd; dk += 32) {
            v16bf a = load_a_frag(Qs + rg * 64 + dk, 64);
            v16bf b = load_b_frag(Ks + sc * 64 + dk, 64);
            s = wmma_bf16(a, b, s);
        }
        // decay mask * scale -> P (bf16 in LDS)
        const int jl = sc + (lane & 15);
        const int jg = j0 + jl;
        const float cj = cg[jg], cj1 = cg[jg + 1];
        for (int r = 0; r < 8; ++r) {
            const int il = rg + r + 8 * hi;
            const int ig = i0 + il;
            const float m = (jg <= ig) ? __expf(ci[r] - cj) : __expf(cj1 - ci1[r]);
            Ps[il * 32 + jl] = f32_to_bf16(s[r] * SCALE * m);
        }
        __syncthreads();

        if (t < 64) {  // row-sum of the 64x32 P panel
            float a = 0.f;
            for (int jj = 0; jj < 32; ++jj) a += bf16_to_f32(Ps[t * 32 + jj]);
            rowsum[t] += a;
        }
        // O += P @ V
        {
            v16bf a  = load_a_frag(Ps + rg * 32, 32);
            v16bf b0 = load_b_frag(Vs + (ocg + 0) * 32, 32);
            v16bf b1 = load_b_frag(Vs + (ocg + 16) * 32, 32);
            o0 = wmma_bf16(a, b0, o0);
            o1 = wmma_bf16(a, b1, o1);
        }
        __syncthreads();
    }

    // normalize; store o as bf16 [B,N,C] (feeds async-staged proj GEMM)
    const int b = bh / Hh, h = bh % Hh;
    const int col = lane & 15;
    for (int ni = 0; ni < 2; ++ni) {
        const v8f& o = ni ? o1 : o0;
        const int d = ocg + ni * 16 + col;
        for (int r = 0; r < 8; ++r) {
            const int il = rg + r + 8 * hi;
            const float rs = rowsum[il] + 1e-6f;
            obf[((size_t)b * Nn + (i0 + il)) * Cc + h * Dd + d] = f32_to_bf16(o[r] / rs);
        }
    }
}

// ================= launch =================
extern "C" void kernel_launch(void* const* d_in, const int* in_sizes, int n_in,
                              void* d_out, int out_size, void* d_ws, size_t ws_size,
                              hipStream_t stream) {
    const float* x     = (const float*)d_in[0];
    const float* Wqkv  = (const float*)d_in[1];
    const float* Wa    = (const float*)d_in[2];
    const float* ba    = (const float*)d_in[3];
    const float* Wproj = (const float*)d_in[4];
    const float* bproj = (const float*)d_in[5];
    float* out = (float*)d_out;

    // workspace layout (bytes, 256-aligned blocks)
    char* ws = (char*)d_ws;
    size_t off = 0;
    float*          cws     = (float*)(ws + off);          off += 98560;      // BH*(N+1) f32
    unsigned short* qkvws   = (unsigned short*)(ws + off); off += 9437184;    // [3,B,H,N,D] bf16
    unsigned short* obfws   = (unsigned short*)(ws + off); off += 3145728;    // [B,N,C] bf16
    unsigned short* xbf     = (unsigned short*)(ws + off); off += 3145728;    // [B*N,C] bf16
    unsigned short* wqkvbf  = (unsigned short*)(ws + off); off += 3538944;    // [3C,C] bf16
    unsigned short* wprojbf = (unsigned short*)(ws + off); off += 1179648;    // [C,C] bf16

    const int nx = Bb * Nn * Cc / 4, nq = 3 * Cc * Cc / 4, np = Cc * Cc / 4;
    cvt_bf16_kernel<<<dim3((nx + 255) / 256), dim3(256), 0, stream>>>(x,     xbf,     nx);
    cvt_bf16_kernel<<<dim3((nq + 255) / 256), dim3(256), 0, stream>>>(Wqkv,  wqkvbf,  nq);
    cvt_bf16_kernel<<<dim3((np + 255) / 256), dim3(256), 0, stream>>>(Wproj, wprojbf, np);

    gates_kernel<<<dim3(BH), dim3(256), 0, stream>>>(x, Wa, ba, cws);
    gemm_kernel<0><<<dim3((3 * Cc) / 64, (Bb * Nn) / 64), dim3(128), 0, stream>>>(
        xbf, wqkvbf, nullptr, qkvws, nullptr);
    qknorm_kernel<<<dim3((2 * BH * Nn) / 8), dim3(256), 0, stream>>>(qkvws);
    attn_kernel<<<dim3(BH, Nn / 64), dim3(256), 0, stream>>>(qkvws, cws, obfws);
    gemm_kernel<1><<<dim3(Cc / 64, (Bb * Nn) / 64), dim3(128), 0, stream>>>(
        obfws, wprojbf, bproj, nullptr, out);
}
